// HybridSTGCN_LSTM_70093866271366
// MI455X (gfx1250) — compile-verified, matrix operand order checked by reading, not run on hardware
//
#include <hip/hip_runtime.h>
#include <hip/hip_bf16.h>

typedef __attribute__((ext_vector_type(16))) _Float16 v16h;
typedef __attribute__((ext_vector_type(8)))  float    v8f;

#define NUM_NODES 2000
#define WINDOW    24
#define NTOT      (NUM_NODES * WINDOW)   // 48000
#define EDGES     768000
#define HID       64
#define GATES     256
#define OUTC      96                      // OUT_C * HORIZON

// branch-free tanh on hardware transcendentals (v_exp_f32 + v_rcp_f32)
__device__ __forceinline__ float fast_tanh(float x) {
    return 2.0f * __builtin_amdgcn_rcpf(1.0f + __expf(-2.0f * x)) - 1.0f;
}

// ---------------------------------------------------------------------------
// utility kernels
// ---------------------------------------------------------------------------
__global__ void k_fill_f32(float* __restrict__ p, float v, int n) {
    int i = blockIdx.x * blockDim.x + threadIdx.x;
    if (i < n) p[i] = v;
}

__global__ void k_deg_accum(const int* __restrict__ dst, float* __restrict__ deg, int e_cnt) {
    int i = blockIdx.x * blockDim.x + threadIdx.x;
    if (i < e_cnt) unsafeAtomicAdd(&deg[dst[i]], 1.0f);
}

__global__ void k_make_dinv(float* __restrict__ deg, int n) {
    int i = blockIdx.x * blockDim.x + threadIdx.x;
    if (i < n) deg[i] = rsqrtf(deg[i]);   // in-place: deg -> deg^{-1/2}
}

// ---------------------------------------------------------------------------
// WMMA GEMM: C[M,N] = A[M,K] @ W(K,N)  (W optionally given transposed [N,K])
// One wave computes a 16xN strip; 8 waves/block -> 128 rows per block.
// W staged raw (f32) into LDS via CDNA5 async global->LDS DMA (ASYNCcnt),
// converted to f16 at fragment-build time. f32 accumulate via
// v_wmma_f32_16x16x32_f16. KB = number of 32-wide K blocks.
// ---------------------------------------------------------------------------
template<int KB, int LDSE>
__global__ __launch_bounds__(256)
void k_gemm_wmma(const float* __restrict__ A, const float* __restrict__ W,
                 const float* __restrict__ bias, float* __restrict__ C,
                 int M, int K, int N, int wt) {
    __shared__ __align__(16) float ldsWf[LDSE];

    const int tid  = threadIdx.x;
    const int wave = tid >> 5;
    const int lane = tid & 31;
    const int halfsel = lane >> 4;       // 0: lanes 0-15, 1: lanes 16-31
    const int basek   = halfsel * 8;
    const int ncol0   = lane & 15;

    // async DMA copy of W (verbatim, f32) into LDS: 16B per issue
    const int total4 = (K * N) >> 2;
    for (int i = tid; i < total4; i += 256) {
        unsigned int lds_off = (unsigned int)(uintptr_t)(&ldsWf[i * 4]);
        unsigned long long ga = (unsigned long long)(uintptr_t)(W + (size_t)i * 4);
        asm volatile("global_load_async_to_lds_b128 %0, %1, off"
                     :: "v"(lds_off), "v"(ga) : "memory");
    }
#if __has_builtin(__builtin_amdgcn_s_wait_asynccnt)
    __builtin_amdgcn_s_wait_asynccnt(0);
#else
    asm volatile("s_wait_asynccnt 0x0" ::: "memory");
#endif
    __syncthreads();

    const int mbase = blockIdx.x * 128 + wave * 16;
    const int mrow  = mbase + (lane & 15);
    const bool mok  = (mrow < M);
    const float* arow = A + (size_t)(mok ? mrow : 0) * K;
    __builtin_prefetch(arow, 0, 0);      // global_prefetch_b8

    // A fragments (ISA 16-bit A 16x32 layout)
    v16h afrag[KB];
#pragma unroll
    for (int kb = 0; kb < KB; ++kb) {
        v16h a;
#pragma unroll
        for (int e = 0; e < 16; ++e) {
            int k = kb * 32 + basek + e + ((e >= 8) ? 8 : 0);
            a[e] = (_Float16)((mok && k < K) ? arow[k] : 0.0f);
        }
        afrag[kb] = a;
    }

    for (int nt = 0; nt < (N >> 4); ++nt) {
        const int ncol = nt * 16 + ncol0;
        v8f c;
#pragma unroll
        for (int r = 0; r < 8; ++r) c[r] = 0.0f;
#pragma unroll
        for (int kb = 0; kb < KB; ++kb) {
            v16h b;
#pragma unroll
            for (int e = 0; e < 16; ++e) {
                int k = kb * 32 + basek + e + ((e >= 8) ? 8 : 0);
                float w = (k < K) ? (wt ? ldsWf[ncol * K + k] : ldsWf[k * N + ncol]) : 0.0f;
                b[e] = (_Float16)w;
            }
            c = __builtin_amdgcn_wmma_f32_16x16x32_f16(false, afrag[kb],
                                                       false, b,
                                                       (short)0, c, false, false);
        }
        const float bv = bias ? bias[ncol] : 0.0f;
#pragma unroll
        for (int r = 0; r < 8; ++r) {
            int m = mbase + r + 8 * halfsel;
            if (m < M) C[(size_t)m * N + ncol] = c[r] + bv;
        }
    }
}

// ---------------------------------------------------------------------------
// GCN edge scatter: acc[dst] += (h@W)[src] * dinv[src]*dinv[dst]
// One thread handles 4 channels of one edge (float4 gather + native f32 atomics,
// all L2-resident: 12MB feature + 12MB accumulator arrays).
// ---------------------------------------------------------------------------
__global__ __launch_bounds__(256)
void k_gcn_scatter(const float* __restrict__ hW, const int* __restrict__ src,
                   const int* __restrict__ dst, const float* __restrict__ dinv,
                   float* __restrict__ acc, int e_cnt) {
    int tid = blockIdx.x * blockDim.x + threadIdx.x;
    int e = tid >> 4;
    int q = tid & 15;
    if (e >= e_cnt) return;
    int s = src[e], d = dst[e];
    float coef = dinv[s] * dinv[d];
    const float4 v = *(const float4*)(hW + (size_t)s * HID + q * 4);
    float* o = acc + (size_t)d * HID + q * 4;
    unsafeAtomicAdd(o + 0, v.x * coef);
    unsafeAtomicAdd(o + 1, v.y * coef);
    unsafeAtomicAdd(o + 2, v.z * coef);
    unsafeAtomicAdd(o + 3, v.w * coef);
}

// out = relu(acc + hW * dinv^2 + b)
__global__ void k_gcn_finalize(const float* __restrict__ hW, const float* __restrict__ acc,
                               const float* __restrict__ dinv, const float* __restrict__ b,
                               float* __restrict__ out, int n) {
    int i = blockIdx.x * blockDim.x + threadIdx.x;
    if (i >= n) return;
    int node = i >> 6, c = i & 63;
    float di = dinv[node];
    float v = acc[i] + hW[i] * di * di + b[c];
    out[i] = fmaxf(v, 0.0f);
}

// ---------------------------------------------------------------------------
// Block-resident LSTM. Block = 4 waves = 16 sequences (nodes).
// Wave w owns gate block w (i,f,g,o). Whh^T WMMA B-fragments live in regs
// for all 24 steps; h lives in LDS as f16 (A-fragment chunks are contiguous
// -> ds_load_b128, no per-step converts), c in registers. x_proj precomputed.
// Activation select (sigmoid vs tanh) is an affine remap of the same
// rcp(1+exp(-s*x)) kernel -> fully branch-free hot loop.
//   xproj layout: [(t*NUM_NODES + node), 256]
//   yout   layout: [(t*NUM_NODES + node), 64]
// ---------------------------------------------------------------------------
__global__ __launch_bounds__(128)
void k_lstm(const float* __restrict__ xproj, const float* __restrict__ Whh,
            const float* __restrict__ bih, const float* __restrict__ bhh,
            float* __restrict__ yout, float* __restrict__ hlast) {
    __shared__ __align__(16) _Float16 lds_hh[16 * 64];
    __shared__ float lds_g[4][16 * 64];

    const int tid  = threadIdx.x;
    const int wave = tid >> 5;
    const int lane = tid & 31;
    const int halfsel = lane >> 4;
    const int basek   = halfsel * 8;
    const int ncol0   = lane & 15;
    const int nodebase = blockIdx.x * 16;

    // activation constants for this wave (wave 2 = cell gate -> tanh)
    const float act_s   = (wave == 2) ? 2.0f : 1.0f;
    const float act_mul = (wave == 2) ? 2.0f : 1.0f;
    const float act_add = (wave == 2) ? -1.0f : 0.0f;

    // B fragments: Whh^T[k=hh, n=g] = Whh[g*64 + k]; this wave's g in [64w,64w+64)
    v16h bfrag[4][2];
#pragma unroll
    for (int nt = 0; nt < 4; ++nt) {
        int g = wave * 64 + nt * 16 + ncol0;
#pragma unroll
        for (int kb = 0; kb < 2; ++kb) {
            v16h b;
#pragma unroll
            for (int e = 0; e < 16; ++e) {
                int k = kb * 32 + basek + e + ((e >= 8) ? 8 : 0);
                b[e] = (_Float16)Whh[g * 64 + k];
            }
            bfrag[nt][kb] = b;
        }
    }

    float cstate[8];
#pragma unroll
    for (int j = 0; j < 8; ++j) cstate[j] = 0.0f;
    for (int i = tid; i < 1024; i += 128) lds_hh[i] = (_Float16)0.0f;
    __syncthreads();

    for (int t = 0; t < WINDOW; ++t) {
        // A fragments from current h (f16 LDS, contiguous 8-half chunks)
        const int mrowA = lane & 15;
        v16h afrag[2];
#pragma unroll
        for (int kb = 0; kb < 2; ++kb) {
            v16h a;
#pragma unroll
            for (int e = 0; e < 16; ++e) {
                int k = kb * 32 + basek + e + ((e >= 8) ? 8 : 0);
                a[e] = lds_hh[mrowA * 64 + k];
            }
            afrag[kb] = a;
        }

        const size_t rowbase = (size_t)t * NUM_NODES + nodebase;
#pragma unroll
        for (int nt = 0; nt < 4; ++nt) {
            const int g = wave * 64 + nt * 16 + ncol0;
            const float bsum = bih[g] + bhh[g];

            // issue the 8 xproj loads first so they hide under the WMMAs
            float xp[8];
#pragma unroll
            for (int r = 0; r < 8; ++r)
                xp[r] = xproj[(rowbase + r + 8 * halfsel) * GATES + g];

            v8f c;
#pragma unroll
            for (int r = 0; r < 8; ++r) c[r] = 0.0f;
            c = __builtin_amdgcn_wmma_f32_16x16x32_f16(false, afrag[0], false, bfrag[nt][0],
                                                       (short)0, c, false, false);
            c = __builtin_amdgcn_wmma_f32_16x16x32_f16(false, afrag[1], false, bfrag[nt][1],
                                                       (short)0, c, false, false);
#pragma unroll
            for (int r = 0; r < 8; ++r) {
                float v = c[r] + xp[r] + bsum;
                float a = __builtin_amdgcn_rcpf(1.0f + __expf(-act_s * v));
                lds_g[wave][(r + 8 * halfsel) * 64 + nt * 16 + ncol0] =
                    fmaf(a, act_mul, act_add);
            }
        }
        __syncthreads();   // gates done; lds_hh fully consumed

        // state update: 128 threads x 8 contiguous elems (idx = m*64+hh)
#pragma unroll
        for (int j = 0; j < 8; ++j) {
            int idx = tid * 8 + j;
            float iv = lds_g[0][idx], fv = lds_g[1][idx];
            float gv = lds_g[2][idx], ov = lds_g[3][idx];
            float cs = fv * cstate[j] + iv * gv;
            cstate[j] = cs;
            float hv = ov * fast_tanh(cs);
            lds_hh[idx] = (_Float16)hv;
            if (yout)  yout[rowbase * 64 + idx] = hv;
            if (hlast && t == WINDOW - 1) hlast[(size_t)nodebase * 64 + idx] = hv;
        }
        __syncthreads();
    }
}

// ---------------------------------------------------------------------------
// launch
// ---------------------------------------------------------------------------
extern "C" void kernel_launch(void* const* d_in, const int* in_sizes, int n_in,
                              void* d_out, int out_size, void* d_ws, size_t ws_size,
                              hipStream_t stream) {
    const float* x    = (const float*)d_in[0];
    const int*   ei   = (const int*)  d_in[1];
    const float* W1   = (const float*)d_in[2];   const float* b1   = (const float*)d_in[3];
    const float* W2   = (const float*)d_in[4];   const float* b2   = (const float*)d_in[5];
    const float* W3   = (const float*)d_in[6];   const float* b3   = (const float*)d_in[7];
    const float* W4   = (const float*)d_in[8];   const float* b4   = (const float*)d_in[9];
    const float* Wih0 = (const float*)d_in[10];  const float* Whh0 = (const float*)d_in[11];
    const float* bih0 = (const float*)d_in[12];  const float* bhh0 = (const float*)d_in[13];
    const float* Wih1 = (const float*)d_in[14];  const float* Whh1 = (const float*)d_in[15];
    const float* bih1 = (const float*)d_in[16];  const float* bhh1 = (const float*)d_in[17];
    const float* Wout = (const float*)d_in[18];  const float* bout = (const float*)d_in[19];
    float* out = (float*)d_out;

    const int* src = ei;
    const int* dst = ei + EDGES;

    // workspace layout (floats)
    float* dinv  = (float*)d_ws;                       // NTOT
    float* gW    = dinv  + NTOT;                       // NTOT*64  (h @ W)
    float* acc   = gW    + (size_t)NTOT * HID;         // NTOT*64  (scatter accum)
    float* hA    = acc   + (size_t)NTOT * HID;         // NTOT*64
    float* hB    = hA    + (size_t)NTOT * HID;         // NTOT*64
    float* xproj = hB    + (size_t)NTOT * HID;         // NTOT*256
    float* ybuf  = xproj + (size_t)NTOT * GATES;       // NTOT*64
    float* hlast = ybuf  + (size_t)NTOT * HID;         // NUM_NODES*64

    const int NH = NTOT * HID;

    // degree -> dinv
    k_fill_f32<<<(NTOT + 255) / 256, 256, 0, stream>>>(dinv, 1.0f, NTOT);
    k_deg_accum<<<(EDGES + 255) / 256, 256, 0, stream>>>(dst, dinv, EDGES);
    k_make_dinv<<<(NTOT + 255) / 256, 256, 0, stream>>>(dinv, NTOT);

    // 4 GCN layers
    const float* hin = x;
    const float* Ws[4] = {W1, W2, W3, W4};
    const float* bs[4] = {b1, b2, b3, b4};
    float* houts[4] = {hA, hB, hA, hB};
    for (int l = 0; l < 4; ++l) {
        if (l == 0)
            k_gemm_wmma<1, 12 * HID><<<(NTOT + 127) / 128, 256, 0, stream>>>(
                hin, Ws[l], nullptr, gW, NTOT, 12, HID, 0);
        else
            k_gemm_wmma<2, HID * HID><<<(NTOT + 127) / 128, 256, 0, stream>>>(
                hin, Ws[l], nullptr, gW, NTOT, HID, HID, 0);
        k_fill_f32<<<(NH + 255) / 256, 256, 0, stream>>>(acc, 0.0f, NH);
        k_gcn_scatter<<<(EDGES * 16 + 255) / 256, 256, 0, stream>>>(gW, src, dst, dinv, acc, EDGES);
        k_gcn_finalize<<<(NH + 255) / 256, 256, 0, stream>>>(gW, acc, dinv, bs[l], houts[l], NH);
        hin = houts[l];
    }

    // LSTM layer 0: x_proj = h4 @ Wih0^T, then recurrent sweep
    k_gemm_wmma<2, HID * GATES><<<(NTOT + 127) / 128, 256, 0, stream>>>(
        hin, Wih0, nullptr, xproj, NTOT, HID, GATES, 1);
    k_lstm<<<NUM_NODES / 16, 128, 0, stream>>>(xproj, Whh0, bih0, bhh0, ybuf, nullptr);

    // LSTM layer 1
    k_gemm_wmma<2, HID * GATES><<<(NTOT + 127) / 128, 256, 0, stream>>>(
        ybuf, Wih1, nullptr, xproj, NTOT, HID, GATES, 1);
    k_lstm<<<NUM_NODES / 16, 128, 0, stream>>>(xproj, Whh1, bih1, bhh1, nullptr, hlast);

    // output head: pred = hlast @ Wout + bout  ->  d_out ([2000,96] == [16000,12])
    k_gemm_wmma<2, HID * OUTC><<<(NUM_NODES + 127) / 128, 256, 0, stream>>>(
        hlast, Wout, bout, out, NUM_NODES, HID, OUTC, 0);
}
